// DeeperIntranodeAggGNN_31619549233248
// MI455X (gfx1250) — matrix operand order repaired and checked by reading, not compiled, not run on hardware
//
#include <hip/hip_runtime.h>

typedef __attribute__((ext_vector_type(16))) _Float16 v16h;
typedef __attribute__((ext_vector_type(8)))  _Float16 v8h;
typedef __attribute__((ext_vector_type(2)))  _Float16 v2h;
typedef __attribute__((ext_vector_type(8)))  float    v8f;

static constexpr int NN = 50000;
static constexpr int EE = 800000;
static constexpr int H  = 64;
static constexpr int LL = 3;

// ---------------- small utility kernels ----------------

__global__ void k_zero(float* p, long n) {
    long i = (long)blockIdx.x * blockDim.x + threadIdx.x;
    if (i < n) p[i] = 0.0f;
}

__global__ void k_deg(const long long* ei, float* deg, int E) {
    int e = blockIdx.x * blockDim.x + threadIdx.x;
    if (e < E) {
        long long d = ei[(long)E + e];
        unsafeAtomicAdd(&deg[d], 1.0f);
    }
}

__global__ void k_invdeg(float* deg, int n) {
    int i = blockIdx.x * blockDim.x + threadIdx.x;
    if (i < n) deg[i] = 1.0f / fmaxf(deg[i], 1.0f);
}

// Convert a 64x64 f32 weight into f16 packed in WMMA B-fragment order:
// pack[((ct4*2 + kki)*32 + lane)*16 + j], where j=2v(+1) maps to
// W[(kki*32 + (lane>>4)*16 + 2v (+1)) * 64 + ct4*16 + (lane&15)].
// A wave's whole B fragment is then one contiguous 32-byte chunk per lane.
__global__ void k_cvt_pack(const float* __restrict__ W, _Float16* __restrict__ dst) {
    int p = blockIdx.x * blockDim.x + threadIdx.x;
    if (p >= 4096) return;
    int j    = p & 15;
    int lane = (p >> 4) & 31;
    int kki  = (p >> 9) & 1;
    int ct4  = p >> 10;
    int v = j >> 1, odd = j & 1, hi = lane >> 4;
    int kb  = kki * 32 + hi * 16 + 2 * v + odd;
    int col = ct4 * 16 + (lane & 15);
    dst[p] = (_Float16)W[kb * 64 + col];
}

// encoder: out[R,64] = in[R,K] @ W[K,64] + b   (tiny K, VALU is fine)
template<int K>
__global__ void k_enc(const float* __restrict__ in, const float* __restrict__ W,
                      const float* __restrict__ b, float* __restrict__ out, int R) {
    long idx = (long)blockIdx.x * blockDim.x + threadIdx.x;
    if (idx >= (long)R * 64) return;
    int r = (int)(idx >> 6), c = (int)(idx & 63);
    float s = b[c];
#pragma unroll
    for (int k = 0; k < K; ++k) s += in[(long)r * K + k] * W[k * 64 + c];
    out[idx] = s;
}

// ---------------- wave32 reduction helper ----------------

__device__ __forceinline__ float wred(float v) {
#pragma unroll
    for (int off = 16; off > 0; off >>= 1) v += __shfl_xor(v, off, 32);
    return v;
}

// ---------------- rowwise LayerNorm + ReLU (wave per row) ----------------

__global__ void k_ln_relu(const float* __restrict__ in, const float* __restrict__ g,
                          const float* __restrict__ b, float* __restrict__ out, int R) {
    int wave = threadIdx.x >> 5, lane = threadIdx.x & 31;
    int r = blockIdx.x * 8 + wave;
    if (r >= R) return;
    size_t base = (size_t)r * 64;
    float v0 = in[base + lane], v1 = in[base + 32 + lane];
    float mu = wred(v0 + v1) * (1.0f / 64.0f);
    float d0 = v0 - mu, d1 = v1 - mu;
    float var = wred(d0 * d0 + d1 * d1) * (1.0f / 64.0f);
    float ri = rsqrtf(var + 1e-5f);
    out[base + lane]      = fmaxf(d0 * ri * g[lane]      + b[lane],      0.0f);
    out[base + 32 + lane] = fmaxf(d1 * ri * g[32 + lane] + b[32 + lane], 0.0f);
}

// ------ fused edge message: gn = relu(LN(g)); m = relu(hn[src]+gn) (f16);
//        agg[dst] += m  (mean scaling applied later via inv_deg) ------

__global__ void k_edge_msg(const float* __restrict__ g, const float* __restrict__ hn,
                           const long long* __restrict__ ei,
                           const float* __restrict__ lg, const float* __restrict__ lb,
                           _Float16* __restrict__ m, float* __restrict__ agg, int E) {
    int wave = threadIdx.x >> 5, lane = threadIdx.x & 31;
    int e = blockIdx.x * 8 + wave;
    if (e >= E) return;
    size_t base = (size_t)e * 64;
    float v0 = g[base + lane], v1 = g[base + 32 + lane];
    float mu = wred(v0 + v1) * (1.0f / 64.0f);
    float d0 = v0 - mu, d1 = v1 - mu;
    float var = wred(d0 * d0 + d1 * d1) * (1.0f / 64.0f);
    float ri = rsqrtf(var + 1e-5f);
    float gn0 = fmaxf(d0 * ri * lg[lane]      + lb[lane],      0.0f);
    float gn1 = fmaxf(d1 * ri * lg[32 + lane] + lb[32 + lane], 0.0f);
    long long s = ei[e], d = ei[(long)E + e];
    float m0 = fmaxf(hn[(size_t)s * 64 + lane]      + gn0, 0.0f);
    float m1 = fmaxf(hn[(size_t)s * 64 + 32 + lane] + gn1, 0.0f);
    m[base + lane]      = (_Float16)m0;
    m[base + 32 + lane] = (_Float16)m1;
    unsafeAtomicAdd(&agg[(size_t)d * 64 + lane],      m0);
    unsafeAtomicAdd(&agg[(size_t)d * 64 + 32 + lane], m1);
}

// ---------------- WMMA GEMM: out[R,64] = ep(A[R,64] @ W[64,64] + bias) ----------------
// AHALF: A is f16 (message tensor). ADD2: A := A + A2*rowscale (mean-agg fused).
// RESID: out = res + acc + bias. RELU: clamp.
// Block: 256 thr = 8 waves = 32 rows x 64 cols; wave -> one 16x16 C tile; K=64 -> 2 WMMAs.
// A staged in LDS as f16 (fragments = 2x ds b128); B pre-packed (1x 32B global per lane);
// C staged back through LDS for a fully-coalesced float4 epilogue.

template<bool AHALF, bool ADD2, bool RELU, bool RESID>
__global__ void k_gemm64(const void* __restrict__ A, const float* __restrict__ A2,
                         const float* __restrict__ rs, const v16h* __restrict__ Wp,
                         const float* __restrict__ bias, const float* __restrict__ res,
                         float* __restrict__ out, int rows) {
    __shared__ float smem[32 * 68];            // C staging (f32); aliased for A staging (f16)
    _Float16* Ah = (_Float16*)smem;            // [32][72] halfs, 16B-aligned rows

    const int tid  = threadIdx.x;
    const int row0 = blockIdx.x * 32;

    // ---- stage A into LDS as f16 (paired writes) ----
    for (int i = tid; i < 32 * 32; i += 256) {           // 1024 half-pairs
        int r = i >> 5, c = (i & 31) * 2;
        int gr = row0 + r;
        v2h pair;
        pair.x = (_Float16)0; pair.y = (_Float16)0;
        if (gr < rows) {
            if (AHALF) {
                pair = *(const v2h*)((const _Float16*)A + (size_t)gr * 64 + c);
            } else {
                const float* Af = (const float*)A;
                float f0 = Af[(size_t)gr * 64 + c];
                float f1 = Af[(size_t)gr * 64 + c + 1];
                if (ADD2) {
                    float sc = rs[gr];
                    f0 += A2[(size_t)gr * 64 + c]     * sc;
                    f1 += A2[(size_t)gr * 64 + c + 1] * sc;
                }
                pair.x = (_Float16)f0; pair.y = (_Float16)f1;
            }
        }
        *(v2h*)(Ah + r * 72 + c) = pair;
    }
    __syncthreads();

    const int wave = tid >> 5, lane = tid & 31;
    const int rt   = (wave >> 2) * 16;     // row tile (0/16)
    const int ct4  = wave & 3;             // col tile index (cols ct4*16)
    const int hi   = lane >> 4;
    const int mrow = rt + (lane & 15);

    v8f acc = {};
#pragma unroll
    for (int kki = 0; kki < 2; ++kki) {
        const int kk = kki * 32;
        // A fragment: lanes0-15 need K {kk+0..15}; lanes16-31 K {kk+16..31} -> two b128s
        const _Float16* arow = Ah + mrow * 72 + kk + hi * 16;
        v8h alo = *(const v8h*)(arow);
        v8h ahi = *(const v8h*)(arow + 8);
        v16h a = __builtin_shufflevector(alo, ahi,
                                         0, 1, 2, 3, 4, 5, 6, 7,
                                         8, 9, 10, 11, 12, 13, 14, 15);
        // B fragment: one contiguous 32B chunk per lane (pre-packed)
        v16h b = Wp[(ct4 * 2 + kki) * 32 + lane];
        acc = __builtin_amdgcn_wmma_f32_16x16x32_f16(
            false, a, false, b, (short)0, acc, false, false);
    }
    __syncthreads();   // everyone done reading Ah

    // ---- scatter C tile to LDS: VGPR r -> M = r + 8*hi, N = lane&15 ----
#pragma unroll
    for (int r = 0; r < 8; ++r) {
        int crow = rt + r + hi * 8;
        smem[crow * 68 + ct4 * 16 + (lane & 15)] = acc[r];
    }
    __syncthreads();

    // ---- coalesced epilogue: thread -> 8 consecutive floats of one row ----
    {
        int r  = tid >> 3;
        int c0 = (tid & 7) * 8;
        int grow = row0 + r;
        if (grow < rows) {
            size_t gbase = (size_t)grow * 64 + c0;
#pragma unroll
            for (int q = 0; q < 2; ++q) {
                float4 cv = *(const float4*)(smem + r * 68 + c0 + q * 4);
                float4 bv = *(const float4*)(bias + c0 + q * 4);
                float4 v;
                v.x = cv.x + bv.x; v.y = cv.y + bv.y;
                v.z = cv.z + bv.z; v.w = cv.w + bv.w;
                if (RESID) {
                    float4 rv = *(const float4*)(res + gbase + q * 4);
                    v.x += rv.x; v.y += rv.y; v.z += rv.z; v.w += rv.w;
                }
                if (RELU) {
                    v.x = fmaxf(v.x, 0.0f); v.y = fmaxf(v.y, 0.0f);
                    v.z = fmaxf(v.z, 0.0f); v.w = fmaxf(v.w, 0.0f);
                }
                *(float4*)(out + gbase + q * 4) = v;
            }
        }
    }
}

// ---------------- decoder: wave per edge, gf-LN computed in-wave ----------------

__global__ void k_decoder(const float* __restrict__ hf, const float* __restrict__ g,
                          const long long* __restrict__ ei,
                          const float* __restrict__ lg, const float* __restrict__ lb,
                          const float* __restrict__ Wd, const float* __restrict__ bd,
                          const float* __restrict__ scale, float* __restrict__ out, int E) {
    int wave = threadIdx.x >> 5, lane = threadIdx.x & 31;
    int e = blockIdx.x * 8 + wave;
    if (e >= E) return;
    size_t base = (size_t)e * 64;
    float v0 = g[base + lane], v1 = g[base + 32 + lane];
    float mu = wred(v0 + v1) * (1.0f / 64.0f);
    float d0 = v0 - mu, d1 = v1 - mu;
    float var = wred(d0 * d0 + d1 * d1) * (1.0f / 64.0f);
    float ri = rsqrtf(var + 1e-5f);
    float gf0 = fmaxf(d0 * ri * lg[lane]      + lb[lane],      0.0f);
    float gf1 = fmaxf(d1 * ri * lg[32 + lane] + lb[32 + lane], 0.0f);
    long long s = ei[e], d = ei[(long)E + e];
    float p = hf[(size_t)s * 64 + lane]      * Wd[lane]
            + hf[(size_t)s * 64 + 32 + lane] * Wd[32 + lane]
            + hf[(size_t)d * 64 + lane]      * Wd[64 + lane]
            + hf[(size_t)d * 64 + 32 + lane] * Wd[96 + lane]
            + gf0 * Wd[128 + lane]
            + gf1 * Wd[160 + lane];
    p = wred(p);
    if (lane == 0) {
        out[(size_t)e * 2]     = fmaxf(p + bd[0], 0.0f);
        out[(size_t)e * 2 + 1] = scale[0];
    }
}

// ---------------- host orchestration ----------------

extern "C" void kernel_launch(void* const* d_in, const int* in_sizes, int n_in,
                              void* d_out, int out_size, void* d_ws, size_t ws_size,
                              hipStream_t stream) {
    const float*     x      = (const float*)d_in[0];
    const float*     ea     = (const float*)d_in[1];
    const long long* ei     = (const long long*)d_in[2];
    const float*     Wn_enc = (const float*)d_in[3];
    const float*     bn_enc = (const float*)d_in[4];
    const float*     We_enc = (const float*)d_in[5];
    const float*     be_enc = (const float*)d_in[6];
    const float*     W0n    = (const float*)d_in[7];
    const float*     b0n    = (const float*)d_in[8];
    const float*     W0e    = (const float*)d_in[9];
    const float*     b0e    = (const float*)d_in[10];
    const float*     ln_g   = (const float*)d_in[11];
    const float*     ln_b   = (const float*)d_in[12];
    const float*     Wnode  = (const float*)d_in[13];
    const float*     bnode  = (const float*)d_in[14];
    const float*     Wedge  = (const float*)d_in[15];
    const float*     bedge  = (const float*)d_in[16];
    const float*     Wdec   = (const float*)d_in[17];
    const float*     bdec   = (const float*)d_in[18];
    const float*     scale  = (const float*)d_in[19];
    float* out = (float*)d_out;

    // workspace carve-up (256B aligned)
    char* ws = (char*)d_ws;
    size_t o = 0;
    auto take = [&](size_t bytes) { size_t r = o; o = (o + bytes + 255) & ~(size_t)255; return r; };
    float*     h    = (float*)(ws + take((size_t)NN * H * 4));
    float*     hn   = (float*)(ws + take((size_t)NN * H * 4));
    float*     agg  = (float*)(ws + take((size_t)NN * H * 4));
    float*     ideg = (float*)(ws + take((size_t)NN * 4));       // deg, then inv in-place
    _Float16*  w16  = (_Float16*)(ws + take((size_t)8 * H * H * 2));
    float*     g    = (float*)(ws + take((size_t)EE * H * 4));
    _Float16*  m    = (_Float16*)(ws + take((size_t)EE * H * 2));

    const int TB = 256;
    // degrees -> inverse
    k_zero<<<(NN + TB - 1) / TB, TB, 0, stream>>>(ideg, NN);
    k_deg<<<(EE + TB - 1) / TB, TB, 0, stream>>>(ei, ideg, EE);
    k_invdeg<<<(NN + TB - 1) / TB, TB, 0, stream>>>(ideg, NN);

    // weights -> f16, pre-packed in B-fragment order:
    // [0]=W0n [1]=W0e [2..4]=Wnode [5..7]=Wedge
    k_cvt_pack<<<16, TB, 0, stream>>>(W0n, w16 + 0 * 4096);
    k_cvt_pack<<<16, TB, 0, stream>>>(W0e, w16 + 1 * 4096);
    for (int i = 0; i < LL; ++i) {
        k_cvt_pack<<<16, TB, 0, stream>>>(Wnode + i * 4096, w16 + (2 + i) * 4096);
        k_cvt_pack<<<16, TB, 0, stream>>>(Wedge + i * 4096, w16 + (5 + i) * 4096);
    }

    // encoders
    k_enc<8><<<((long)NN * 64 + TB - 1) / TB, TB, 0, stream>>>(x, Wn_enc, bn_enc, h, NN);
    k_enc<4><<<((long)EE * 64 + TB - 1) / TB, TB, 0, stream>>>(ea, We_enc, be_enc, g, EE);

    const int NB = (NN + 31) / 32;   // 1563
    const int EB = (EE + 31) / 32;   // 25000

    // layer 0: local transforms with relu (in-place safe: block reads own rows first)
    k_gemm64<false, false, true, false><<<NB, TB, 0, stream>>>(
        h, nullptr, nullptr, (const v16h*)(w16 + 0 * 4096), b0n, nullptr, h, NN);
    k_gemm64<false, false, true, false><<<EB, TB, 0, stream>>>(
        g, nullptr, nullptr, (const v16h*)(w16 + 1 * 4096), b0e, nullptr, g, EE);

    // res+ layers
    for (int i = 0; i < LL; ++i) {
        k_ln_relu<<<(NN + 7) / 8, TB, 0, stream>>>(h, ln_g + i * 64, ln_b + i * 64, hn, NN);
        k_zero<<<((long)NN * 64 + TB - 1) / TB, TB, 0, stream>>>(agg, (long)NN * 64);
        k_edge_msg<<<(EE + 7) / 8, TB, 0, stream>>>(
            g, hn, ei, ln_g + i * 64, ln_b + i * 64, m, agg, EE);
        // h = h + (hn + agg*inv_deg) @ Wnode[i] + bnode[i]
        k_gemm64<false, true, false, true><<<NB, TB, 0, stream>>>(
            hn, agg, ideg, (const v16h*)(w16 + (2 + i) * 4096), bnode + i * 64, h, h, NN);
        // g = g + m @ Wedge[i] + bedge[i]
        k_gemm64<true, false, false, true><<<EB, TB, 0, stream>>>(
            m, nullptr, nullptr, (const v16h*)(w16 + (5 + i) * 4096), bedge + i * 64, g, g, EE);
    }

    // final norm+act for nodes (layer-0 params), then fused decoder
    k_ln_relu<<<(NN + 7) / 8, TB, 0, stream>>>(h, ln_g, ln_b, hn, NN);
    k_decoder<<<(EE + 7) / 8, TB, 0, stream>>>(
        hn, g, ei, ln_g, ln_b, Wdec, bdec, scale, out, EE);
}
